// GATConv_47742856463110
// MI455X (gfx1250) — matrix operand order repaired and checked by reading, not compiled, not run on hardware
//
#include <hip/hip_runtime.h>
#include <hip/hip_bf16.h>

#define N_NODES 4096
#define IN_FEAT 256
#define OUT_FEAT 32
#define NHEAD 8
#define NEG_SLOPE 0.2f
#define MAX_NBR 1024

typedef __attribute__((ext_vector_type(16))) __bf16 v16bf;
typedef __attribute__((ext_vector_type(8)))  float  v8f;

// Native bf16 convert: fptrunc lowers to the hardware cvt on gfx1250
// (or to the standard RNE integer expansion if not — never worse).
static __device__ __forceinline__ __bf16 f2bf(float f) { return (__bf16)f; }

// -------- Stage 1: Wh[h] = X @ W[h]  (bf16 WMMA, fp32 accumulate) --------
// grid (64, 2, 8), block 128 (4 waves); each wave owns one 16x16 output tile.
__global__ void gat_gemm_wmma(const float* __restrict__ x,
                              const float* __restrict__ W,
                              float* __restrict__ Wh) {
  const int lane = threadIdx.x & 31;
  const int wv   = threadIdx.x >> 5;
  const int m0   = (blockIdx.x * 4 + wv) * 16;      // node tile
  const int n0   = blockIdx.y * 16;                 // feature tile
  const int h    = blockIdx.z;

  const float* xrow = x + (size_t)(m0 + (lane & 15)) * IN_FEAT;
  const float* wmat = W + (size_t)h * IN_FEAT * OUT_FEAT;
  const int nB   = n0 + (lane & 15);
  const int aKof = (lane >> 4) * 8;                 // A K sub-offset per half-wave
  const int bKof = (lane >> 4) * 8;                 // B K sub-offset per half-wave

  v8f c = {};
  #pragma unroll
  for (int k0 = 0; k0 < IN_FEAT; k0 += 32) {
    v16bf a, b;
    #pragma unroll
    for (int e = 0; e < 8; ++e) {
      a[e]     = f2bf(xrow[k0 + aKof + e]);
      a[e + 8] = f2bf(xrow[k0 + 16 + aKof + e]);
    }
    #pragma unroll
    for (int e = 0; e < 8; ++e) {
      b[e]     = f2bf(wmat[(size_t)(k0 + bKof + e) * OUT_FEAT + nB]);
      b[e + 8] = f2bf(wmat[(size_t)(k0 + 16 + bKof + e) * OUT_FEAT + nB]);
    }
    c = __builtin_amdgcn_wmma_f32_16x16x32_bf16(false, a, false, b,
                                                (short)0, c, false, false);
  }
  const int mo = (lane >> 4) * 8;
  #pragma unroll
  for (int v = 0; v < 8; ++v)
    Wh[((size_t)h * N_NODES + (m0 + v + mo)) * OUT_FEAT + nB] = c[v];
}

// -------- Stage 1b: attention logits e_src/e_dst = Wh . a --------
__global__ void gat_logits(const float* __restrict__ Wh,
                           const float* __restrict__ a_src,
                           const float* __restrict__ a_dst,
                           float* __restrict__ e_src,
                           float* __restrict__ e_dst) {
  int idx = blockIdx.x * blockDim.x + threadIdx.x;  // idx = h*N + n
  if (idx >= NHEAD * N_NODES) return;
  int h = idx >> 12;                                // N_NODES == 4096
  const float* wrow = Wh + (size_t)idx * OUT_FEAT;
  const float* as = a_src + h * OUT_FEAT;
  const float* ad = a_dst + h * OUT_FEAT;
  float s1 = 0.f, s2 = 0.f;
  #pragma unroll
  for (int f = 0; f < OUT_FEAT; ++f) {
    float w = wrow[f];
    s1 += w * as[f];
    s2 += w * ad[f];
  }
  e_src[idx] = s1;
  e_dst[idx] = s2;
}

// -------- Stage 2: sparse masked softmax + aggregation --------
// One block per node i.
// Phase A: thread t loads 16 adjacency bytes with ONE b128 load, builds a
//          16-bit hit mask (incl. self-loop), and a block-wide Hillis-Steele
//          scan places hits into an ascending, deterministic LDS list.
// Phase B: one wave per head (8 waves), lane = output feature, online softmax.
__global__ void gat_aggregate(const unsigned char* __restrict__ adj,
                              const float* __restrict__ Wh,
                              const float* __restrict__ e_src,
                              const float* __restrict__ e_dst,
                              const float* __restrict__ bias,
                              float* __restrict__ out) {
  __shared__ unsigned short nbr[MAX_NBR];
  __shared__ int scan[256];

  const int i    = blockIdx.x;
  const int tid  = threadIdx.x;
  const int lane = tid & 31;
  const int wid  = tid >> 5;

  // one 16-byte load per thread covers the whole 4096-byte row
  const uint4* arow4 = (const uint4*)(adj + (size_t)i * N_NODES);
  uint4 v = arow4[tid];
  const int base = tid * 16;

  unsigned hm = 0;
  unsigned wds[4] = {v.x, v.y, v.z, v.w};
  #pragma unroll
  for (int w = 0; w < 4; ++w) {
    #pragma unroll
    for (int b = 0; b < 4; ++b) {
      int idx = w * 4 + b;
      unsigned byte = (wds[w] >> (8 * b)) & 0xFFu;
      if (byte != 0u || (base + idx) == i) hm |= (1u << idx);
    }
  }
  const int cnt = __popc(hm);
  scan[tid] = cnt;
  __syncthreads();

  // Hillis-Steele inclusive scan over 256 counts (8 steps)
  #pragma unroll
  for (int off = 1; off < 256; off <<= 1) {
    int t   = scan[tid];
    int add = (tid >= off) ? scan[tid - off] : 0;
    __syncthreads();
    scan[tid] = t + add;
    __syncthreads();
  }
  int pos   = scan[tid] - cnt;        // exclusive prefix
  int total = scan[255];
  if (total > MAX_NBR) total = MAX_NBR;

  // emit hits in ascending j order (deterministic)
  unsigned rem = hm;
  while (rem) {
    int b = __ffs(rem) - 1;
    rem &= rem - 1u;
    if (pos < MAX_NBR) nbr[pos] = (unsigned short)(base + b);
    ++pos;
  }
  __syncthreads();

  // ---- Phase B: one wave per head, online softmax over neighbor list ----
  const int h = wid;
  const float ei = e_src[h * N_NODES + i];
  const float* edst = e_dst + (size_t)h * N_NODES;
  const float* whh  = Wh + (size_t)h * N_NODES * OUT_FEAT;

  float m = -INFINITY, s = 0.f, acc = 0.f;
  for (int t = 0; t < total; ++t) {
    int j = nbr[t];
    float e = ei + edst[j];
    e = (e >= 0.f) ? e : NEG_SLOPE * e;             // LeakyReLU
    if (e > m) {                                    // online softmax rescale
      float sc = __expf(m - e);
      s *= sc; acc *= sc; m = e;
    }
    float p = __expf(e - m);
    s += p;
    acc += p * whh[(size_t)j * OUT_FEAT + lane];    // coalesced 32-float row
  }
  out[(size_t)i * (NHEAD * OUT_FEAT) + h * OUT_FEAT + lane] =
      acc / s + bias[h * OUT_FEAT + lane];
}

extern "C" void kernel_launch(void* const* d_in, const int* in_sizes, int n_in,
                              void* d_out, int out_size, void* d_ws, size_t ws_size,
                              hipStream_t stream) {
  (void)in_sizes; (void)n_in; (void)out_size; (void)ws_size;
  const float*         x     = (const float*)d_in[0];
  const unsigned char* adj   = (const unsigned char*)d_in[1];
  const float*         W     = (const float*)d_in[2];
  const float*         a_src = (const float*)d_in[3];
  const float*         a_dst = (const float*)d_in[4];
  const float*         bias  = (const float*)d_in[5];
  float* out = (float*)d_out;

  float* Wh    = (float*)d_ws;                                  // 8*4096*32 f32 = 4 MB
  float* e_src = Wh + (size_t)NHEAD * N_NODES * OUT_FEAT;       // 32768 f32
  float* e_dst = e_src + (size_t)NHEAD * N_NODES;               // 32768 f32

  dim3 g1(N_NODES / 16 / 4, OUT_FEAT / 16, NHEAD);              // (64, 2, 8)
  gat_gemm_wmma<<<g1, 128, 0, stream>>>(x, W, Wh);
  gat_logits<<<(NHEAD * N_NODES) / 256, 256, 0, stream>>>(Wh, a_src, a_dst,
                                                          e_src, e_dst);
  gat_aggregate<<<N_NODES, 256, 0, stream>>>(adj, Wh, e_src, e_dst, bias, out);
}
